// GaussianMoELayer_67491116089918
// MI455X (gfx1250) — matrix operand order repaired and checked by reading, not compiled, not run on hardware
//
#include <hip/hip_runtime.h>
#include <hip/hip_bf16.h>
#include <math.h>

// Shapes (fixed by the reference)
#define NTOK 8192
#define DIM  1024
#define HID  4096
#define OUTF 1024
#define NEXP 8
#define CAP  8192          // per-expert pair capacity (worst case)

// d_out flat layout (float): final_output | log_probs | weights | top_idx
#define OUT_LP  (NTOK * OUTF)            // 8388608
#define OUT_W   (OUT_LP + NTOK * NEXP)   // 8454144
#define OUT_IDX (OUT_W + NTOK * NEXP)    // 8519680

typedef __attribute__((ext_vector_type(16))) _Float16 v16h;
typedef __attribute__((ext_vector_type(8)))  _Float16 v8h;
typedef __attribute__((ext_vector_type(8)))  float    v8f;
typedef __attribute__((ext_vector_type(4)))  int      v4i;

#define AS1 __attribute__((address_space(1)))
#define AS3 __attribute__((address_space(3)))

#if __has_builtin(__builtin_amdgcn_global_load_async_to_lds_b128)
#define HAVE_ASYNC_LDS 1
#else
#define HAVE_ASYNC_LDS 0
#endif

union AFrag { v16h v; v8h h[2]; };

// Assemble a 16-bit WMMA A/B fragment from two 16B chunks 16 halves apart
// (CDNA5 16-bit A layout: lane<16 -> K {0..7,16..23}, lane>=16 -> {8..15,24..31}).
__device__ __forceinline__ v16h load_frag_split(const _Float16* p) {
    AFrag a;
    a.h[0] = *(const v8h*)(p);
    a.h[1] = *(const v8h*)(p + 16);
    return a.v;
}

// wave32 butterfly reduce via ds_swizzle (group-of-32: xor in [14:10], and=0x1f)
__device__ __forceinline__ float wave_reduce_add(float v) {
    v += __int_as_float(__builtin_amdgcn_ds_swizzle(__float_as_int(v), 0x401F)); // xor 16
    v += __int_as_float(__builtin_amdgcn_ds_swizzle(__float_as_int(v), 0x201F)); // xor 8
    v += __int_as_float(__builtin_amdgcn_ds_swizzle(__float_as_int(v), 0x101F)); // xor 4
    v += __int_as_float(__builtin_amdgcn_ds_swizzle(__float_as_int(v), 0x081F)); // xor 2
    v += __int_as_float(__builtin_amdgcn_ds_swizzle(__float_as_int(v), 0x041F)); // xor 1
    return v;
}

// ---------------- 1) inv_sigma = exp(-log_sigma), lsum[e] = sum(log_sigma) ----------------
__global__ __launch_bounds__(256) void prep_sigma_kernel(
    const float* __restrict__ log_sigmas, float* __restrict__ inv_sigma,
    float* __restrict__ lsum) {
    int e = threadIdx.x >> 5, lane = threadIdx.x & 31;
    float s = 0.f;
    for (int j = 0; j < DIM / 32; ++j) {
        int d = lane + j * 32;
        float ls = log_sigmas[e * DIM + d];
        inv_sigma[e * DIM + d] = expf(-ls);
        s += ls;
    }
    s = wave_reduce_add(s);
    if (lane == 0) lsum[e] = s;
}

// ---------------- 2) tiled transpose + f32->f16: dst[e][c][r] = src[e][r][c] ----------------
__global__ __launch_bounds__(256) void transpose_to_f16_kernel(
    const float* __restrict__ src, _Float16* __restrict__ dst, int R, int C) {
    __shared__ float tile[32][33];
    int e = blockIdx.z;
    int c0 = blockIdx.x * 32, r0 = blockIdx.y * 32;
    int tx = threadIdx.x, ty = threadIdx.y;   // blockDim = (32, 8)
    const float* s = src + (size_t)e * R * C;
    _Float16*    d = dst + (size_t)e * R * C;
#pragma unroll
    for (int i = 0; i < 4; ++i)
        tile[ty + i * 8][tx] = s[(size_t)(r0 + ty + i * 8) * C + c0 + tx];
    __syncthreads();
#pragma unroll
    for (int i = 0; i < 4; ++i)
        d[(size_t)(c0 + ty + i * 8) * R + r0 + tx] = (_Float16)tile[tx][ty + i * 8];
}

// ---------------- 3) x -> f16 ----------------
__global__ __launch_bounds__(256) void convert_x_kernel(
    const float* __restrict__ x, _Float16* __restrict__ xh, int n) {
    int i = blockIdx.x * 256 + threadIdx.x;
    if (i < n) xh[i] = (_Float16)x[i];
}

// ---------------- 4) routing: log-probs, softmax, top-2 ----------------
__global__ __launch_bounds__(256) void routing_kernel(
    const float* __restrict__ x, const float* __restrict__ mus,
    const float* __restrict__ inv_sigma, const float* __restrict__ lsum,
    float* __restrict__ out, int* __restrict__ ridx, float* __restrict__ rww) {
    int wave = threadIdx.x >> 5, lane = threadIdx.x & 31;
    int n = blockIdx.x * 8 + wave;
    const float* xr = x + (size_t)n * DIM;
    float acc[NEXP];
#pragma unroll
    for (int e = 0; e < NEXP; ++e) acc[e] = 0.f;
    for (int j = 0; j < DIM / 32; ++j) {
        int d = lane + j * 32;
        float xv = xr[d];
#pragma unroll
        for (int e = 0; e < NEXP; ++e) {
            float diff = (xv - mus[e * DIM + d]) * inv_sigma[e * DIM + d];
            acc[e] += diff * diff;
        }
    }
#pragma unroll
    for (int e = 0; e < NEXP; ++e) acc[e] = wave_reduce_add(acc[e]);
    if (lane == 0) {
        float lp[NEXP], w[NEXP];
        float m = -3.4e38f;
        for (int e = 0; e < NEXP; ++e) {
            lp[e] = -0.5f * acc[e] - lsum[e];
            if (lp[e] > m) m = lp[e];
        }
        float s = 0.f;
        for (int e = 0; e < NEXP; ++e) { w[e] = expf(lp[e] - m); s += w[e]; }
        float inv = 1.f / s;
        for (int e = 0; e < NEXP; ++e) {
            w[e] *= inv;
            out[OUT_LP + n * NEXP + e] = lp[e];
            out[OUT_W  + n * NEXP + e] = w[e];
        }
        int i0 = 0;
        for (int e = 1; e < NEXP; ++e) if (w[e] > w[i0]) i0 = e;
        int i1 = (i0 == 0) ? 1 : 0;
        for (int e = 0; e < NEXP; ++e) if (e != i0 && w[e] > w[i1]) i1 = e;
        float s2 = w[i0] + w[i1] + 1e-9f;
        out[OUT_IDX + n * 2 + 0] = (float)i0;
        out[OUT_IDX + n * 2 + 1] = (float)i1;
        ridx[n * 2 + 0] = i0;  rww[n * 2 + 0] = w[i0] / s2;
        ridx[n * 2 + 1] = i1;  rww[n * 2 + 1] = w[i1] / s2;
    }
}

// ---------------- 5) deterministic ballot-compaction per expert ----------------
__global__ __launch_bounds__(256) void build_lists_kernel(
    const int* __restrict__ ridx, const float* __restrict__ rww,
    int* __restrict__ pair_pd, float* __restrict__ pair_w, int* __restrict__ counts) {
    int e = threadIdx.x >> 5, lane = threadIdx.x & 31;
    unsigned cnt = 0;
    unsigned lanemask = (1u << lane) - 1u;
    for (int base = 0; base < NTOK; base += 32) {
        int tok = base + lane;
        int i0 = ridx[tok * 2], i1 = ridx[tok * 2 + 1];
        bool m0 = (i0 == e), m1 = (i1 == e);
        bool m = m0 || m1;
        unsigned bal = __builtin_amdgcn_ballot_w32(m);
        if (m) {
            unsigned pos = cnt + (unsigned)__builtin_popcount(bal & lanemask);
            pair_pd[e * CAP + pos] = tok * 2 + (m0 ? 0 : 1);
            pair_w [e * CAP + pos] = m0 ? rww[tok * 2] : rww[tok * 2 + 1];
        }
        cnt += (unsigned)__builtin_popcount(bal);
    }
    if (lane == 0) counts[e] = (int)cnt;
}

// ---------------- 6) fused two-GEMM expert MLP with f16 WMMA ----------------
#define XTS 1032   // padded x-tile row stride (halves): 2064B = 4 banks shift/row
#define HBS 136    // padded h-block row stride (halves)
#define HBLK 128   // H chunk

__global__ __launch_bounds__(256) void moe_mlp_kernel(
    const _Float16* __restrict__ xh, const _Float16* __restrict__ W1t,
    const _Float16* __restrict__ W2t, const float* __restrict__ b1,
    const float* __restrict__ b2, const int* __restrict__ pair_pd,
    const float* __restrict__ pair_w, const int* __restrict__ counts,
    float* __restrict__ y_top) {
    __shared__ __align__(16) _Float16 xt[16 * XTS];
    __shared__ __align__(16) _Float16 hblk[2][16 * HBS];  // double-buffered
    __shared__ int   s_pd[16];
    __shared__ float s_w[16];

    int e = blockIdx.y;
    int cnt = counts[e];
    int row0 = blockIdx.x << 4;
    if (row0 >= cnt) return;                 // uniform: EXEC stays all-ones for WMMA

    int tid = threadIdx.x, wave = tid >> 5, lane = tid & 31;
    int khalf = lane >> 4, nlane = lane & 15;
    int acol0 = khalf * 8;

    if (tid < 16) {
        int r = row0 + tid;
        s_pd[tid] = (r < cnt) ? pair_pd[e * CAP + r] : -1;
        s_w[tid]  = (r < cnt) ? pair_w[e * CAP + r] : 0.f;
    }
    __syncthreads();

    {   // stage x tile: 16 rows x 1024 halves; 128 bytes per thread
        int row = tid >> 4;
        int cb = (tid & 15) * 64;
        int pd = s_pd[row];
        if (pd >= 0) {
#if HAVE_ASYNC_LDS
            // gfx1250 async copy: memory -> LDS, no VGPR round-trip (ASYNCcnt)
            AS1 v4i* src = (AS1 v4i*)(xh + ((size_t)(pd >> 1) << 10) + cb);
            AS3 v4i* dst = (AS3 v4i*)&xt[row * XTS + cb];
#pragma unroll
            for (int i = 0; i < 8; ++i)
                __builtin_amdgcn_global_load_async_to_lds_b128(
                    src + i, dst + i, 0, 0);
#else
            const _Float16* src = xh + ((size_t)(pd >> 1) << 10) + cb;
#pragma unroll
            for (int i = 0; i < 8; ++i)
                *(v8h*)&xt[row * XTS + cb + i * 8] = *(const v8h*)(src + i * 8);
#endif
        } else {
            _Float16 z = (_Float16)0.f;
            v8h zv = { z, z, z, z, z, z, z, z };
#pragma unroll
            for (int i = 0; i < 8; ++i)
                *(v8h*)&xt[row * XTS + cb + i * 8] = zv;
        }
    }
#if HAVE_ASYNC_LDS
#if __has_builtin(__builtin_amdgcn_s_wait_asynccnt)
    __builtin_amdgcn_s_wait_asynccnt(0);
#else
    asm volatile("s_wait_asynccnt 0" ::: "memory");
#endif
#endif
    __syncthreads();

    v8f yacc[8];
#pragma unroll
    for (int t = 0; t < 8; ++t) yacc[t] = {};

    const _Float16* W1te = W1t + ((size_t)e << 22);  // [H][K]
    const _Float16* W2te = W2t + ((size_t)e << 22);  // [O][H]
    const _Float16* arow = &xt[nlane * XTS] + acol0;

    int pb = 0;
    for (int hb = 0; hb < HID; hb += HBLK, pb ^= 1) {
        // GEMM1: this wave computes h tile cols [hb + wave*16, +16), K = 1024
        v8f hacc = {};
        int hcol = hb + (wave << 4) + nlane;
        const _Float16* bp = W1te + ((size_t)hcol << 10) + (khalf << 3);
#pragma unroll 4
        for (int kb = 0; kb < DIM; kb += 32) {
            v16h A = load_frag_split(arow + kb);
            v16h B = load_frag_split(bp + kb);
            hacc = __builtin_amdgcn_wmma_f32_16x16x32_f16(
                false, A, false, B, (short)0, hacc, false, false);
        }
        // bias + exact GELU, stash h block in LDS as f16
        float b1v = b1[(e << 12) + hcol];
        _Float16* hout = &hblk[pb][(wave << 4) + nlane];
#pragma unroll
        for (int r = 0; r < 8; ++r) {
            float hv = hacc[r] + b1v;
            hv = 0.5f * hv * (1.0f + erff(hv * 0.70710678118654752f));
            hout[(r + acol0) * HBS] = (_Float16)hv;
        }
        __syncthreads();   // single barrier per H-chunk (double buffer covers WAR)

        // warm caches for next H-chunk's W1 columns (global_prefetch_b8)
        if (hb + HBLK < HID)
            __builtin_prefetch(bp + ((size_t)HBLK << 10), 0, 3);

        // GEMM2 partial: this wave owns out cols [wave*128, +128), K = 128
        const _Float16* ar2 = &hblk[pb][nlane * HBS] + acol0;
#pragma unroll
        for (int t = 0; t < 8; ++t) {
            int ocol = (wave << 7) + (t << 4) + nlane;
            const _Float16* bp2 = W2te + ((size_t)ocol << 12) + hb + (khalf << 3);
#pragma unroll
            for (int kk = 0; kk < 4; ++kk) {
                v16h A = load_frag_split(ar2 + kk * 32);
                v16h B = load_frag_split(bp2 + kk * 32);
                yacc[t] = __builtin_amdgcn_wmma_f32_16x16x32_f16(
                    false, A, false, B, (short)0, yacc[t], false, false);
            }
        }
    }

    // epilogue: y_top[pair][ocol] = w * (y + b2)
#pragma unroll
    for (int t = 0; t < 8; ++t) {
        int ocol = (wave << 7) + (t << 4) + nlane;
        float b2v = b2[(e << 10) + ocol];
#pragma unroll
        for (int r = 0; r < 8; ++r) {
            int rowm = r + acol0;
            int pd = s_pd[rowm];
            if (pd >= 0)
                y_top[((size_t)pd << 10) + ocol] = s_w[rowm] * (yacc[t][r] + b2v);
        }
    }
}

// ---------------- 7) deterministic combine of the two expert slots ----------------
__global__ __launch_bounds__(256) void combine_kernel(
    const float* __restrict__ y_top, float* __restrict__ out) {
    int i = blockIdx.x * 256 + threadIdx.x;       // i in [0, NTOK*OUTF)
    int n = i >> 10;
    size_t base = (size_t)i + ((size_t)n << 10);  // (2n)*1024 + o
    out[i] = y_top[base] + y_top[base + 1024];
}

extern "C" void kernel_launch(void* const* d_in, const int* in_sizes, int n_in,
                              void* d_out, int out_size, void* d_ws, size_t ws_size,
                              hipStream_t stream) {
    const float* x          = (const float*)d_in[0];
    const float* mus        = (const float*)d_in[1];
    const float* log_sigmas = (const float*)d_in[2];
    const float* W1         = (const float*)d_in[3];
    const float* b1         = (const float*)d_in[4];
    const float* W2         = (const float*)d_in[5];
    const float* b2         = (const float*)d_in[6];
    float* out = (float*)d_out;
    (void)in_sizes; (void)n_in; (void)out_size; (void)ws_size;

    // workspace carve-up (~219 MB total)
    char* ws = (char*)d_ws;
    size_t off = 0;
    auto carve = [&](size_t bytes) -> char* {
        char* p = ws + off;
        off += (bytes + 255) & ~(size_t)255;
        return p;
    };
    _Float16* W1t = (_Float16*)carve((size_t)NEXP * HID * DIM * 2);   // 64 MB [E][H][K]
    _Float16* W2t = (_Float16*)carve((size_t)NEXP * OUTF * HID * 2);  // 64 MB [E][O][H]
    _Float16* xh  = (_Float16*)carve((size_t)NTOK * DIM * 2);         // 16 MB
    float* inv_sigma = (float*)carve((size_t)NEXP * DIM * 4);
    float* lsum      = (float*)carve(64);
    int*   ridx      = (int*)  carve((size_t)NTOK * 2 * 4);
    float* rww       = (float*)carve((size_t)NTOK * 2 * 4);
    int*   pair_pd   = (int*)  carve((size_t)NEXP * CAP * 4);
    float* pair_w    = (float*)carve((size_t)NEXP * CAP * 4);
    int*   counts    = (int*)  carve(64);
    float* y_top     = (float*)carve((size_t)NTOK * 2 * OUTF * 4);    // 64 MB

    prep_sigma_kernel<<<1, 256, 0, stream>>>(log_sigmas, inv_sigma, lsum);
    transpose_to_f16_kernel<<<dim3(HID / 32, DIM / 32, NEXP), dim3(32, 8), 0, stream>>>(
        W1, W1t, DIM, HID);
    transpose_to_f16_kernel<<<dim3(OUTF / 32, HID / 32, NEXP), dim3(32, 8), 0, stream>>>(
        W2, W2t, HID, OUTF);
    convert_x_kernel<<<(NTOK * DIM) / 256, 256, 0, stream>>>(x, xh, NTOK * DIM);
    routing_kernel<<<NTOK / 8, 256, 0, stream>>>(x, mus, inv_sigma, lsum, out, ridx, rww);
    build_lists_kernel<<<1, 256, 0, stream>>>(ridx, rww, pair_pd, pair_w, counts);
    moe_mlp_kernel<<<dim3(CAP / 16, NEXP), 256, 0, stream>>>(
        xh, W1t, W2t, b1, b2, pair_pd, pair_w, counts, y_top);
    combine_kernel<<<(NTOK * OUTF) / 256, 256, 0, stream>>>(y_top, out);
}
